// PointerNetwork_26113401160000
// MI455X (gfx1250) — compile-verified, matrix operand order checked by reading, not run on hardware
//
#include <hip/hip_runtime.h>
#include <math.h>

// Problem constants (from reference)
#define EE 256    // embedding dim
#define WD 256    // attention width W
#define LL 512    // sequence length
#define TT 64     // decoder steps
#define HH 512    // hidden
#define BB 64     // batch
#define G4 2048   // 4*H (gate width)

typedef __attribute__((ext_vector_type(16))) _Float16 v16h;
typedef __attribute__((ext_vector_type(8)))  _Float16 v8h;
typedef __attribute__((ext_vector_type(8)))  float    v8f;

// ---- WMMA fragment helpers (v_wmma_f32_16x16x32_f16 layouts, wave32) ----
// A (16x32 f16, row-major source): lane m=lane&15; lanes<16 hold K 0-7 & 16-23,
// lanes>=16 hold K 8-15 & 24-31  ->  two contiguous 16B loads per fragment.
static __device__ inline v16h frag_from_row(const _Float16* p) {
  v8h lo = *(const v8h*)(p);
  v8h hi = *(const v8h*)(p + 16);
  v16h a;
#pragma unroll
  for (int i = 0; i < 8; ++i) { a[i] = lo[i]; a[i + 8] = hi[i]; }
  return a;
}

static __device__ inline float sigm(float x) { return 1.0f / (1.0f + __expf(-x)); }

// ---- Pack a weight matrix Wsrc[N][K] (f32, used as B = W^T, K x N) into
// WMMA-B fragment order: pack[((nt*KT + kt)*32 + lane)*16 + e] so the GEMM
// reads one contiguous 32B v16h per tile per lane. ----
__global__ void k_pack_b(const float* __restrict__ Wsrc, _Float16* __restrict__ pack,
                         int N, int K) {
  int idx = blockIdx.x * blockDim.x + threadIdx.x;
  if (idx >= N * K) return;
  int e    = idx & 15;
  int lane = (idx >> 4) & 31;
  int rest = idx >> 9;
  int KT   = K >> 5;
  int kt   = rest % KT;
  int nt   = rest / KT;
  int n = nt * 16 + (lane & 15);
  int k = kt * 32 + (e & 7) + ((lane >> 4) & 1) * 8 + (e >> 3) * 16;
  pack[idx] = (_Float16)Wsrc[(size_t)n * K + k];
}

__global__ void k_add2(const float* __restrict__ a, const float* __restrict__ b,
                       float* __restrict__ o, int n) {
  int i = blockIdx.x * blockDim.x + threadIdx.x;
  if (i < n) o[i] = a[i] + b[i];
}

__global__ void k_f32_to_f16(const float* __restrict__ s, _Float16* __restrict__ d, int n) {
  int i = blockIdx.x * blockDim.x + threadIdx.x;
  if (i < n) d[i] = (_Float16)s[i];
}

// ---- Fused LSTM step: gate GEMM (WMMA) + bias + cell nonlinearity.
// Grid: 32 blocks (one per 16-wide hidden slice) x 512 threads (16 waves).
// Wave w: m-tile = w&3 (batch 64 = 4 tiles), gate quadrant q = w>>2 (i,f,g,o).
// Each wave computes one 16x16 gate tile, stages it in LDS; after syncthreads
// the block applies the LSTM cell for its 64x16 slice of (c, h).
// h is double-buffered across steps (h_in / h_out) to avoid cross-block RAW.
__global__ void k_lstm_step(const _Float16* __restrict__ h_in,    // [64,512] f16
                            _Float16* __restrict__ h_out,         // [64,512] f16
                            float* __restrict__ c,                // [64,512] f32 in/out
                            const float* __restrict__ emb,        // [V,256] or null
                            const int* __restrict__ ids, int t,   // [B,L] or null
                            const _Float16* __restrict__ packWih, // packed, K=256 (or null)
                            const _Float16* __restrict__ packWhh, // packed, K=512
                            const float* __restrict__ bias,       // [2048] (bih+bhh)
                            _Float16* __restrict__ enc_out,       // f16 enc_states row or null
                            float* __restrict__ h32_out) {        // f32 h (c0 capture) or null
  __shared__ float gbuf[4][64][17];  // padded: no LDS bank conflicts on read
  int wv   = threadIdx.x >> 5;   // 0..15
  int lane = threadIdx.x & 31;
  int mt = wv & 3;               // batch tile
  int q  = wv >> 2;              // gate quadrant (i,f,g,o)
  int j0 = blockIdx.x * 16;      // hidden slice base
  int nt = q * 32 + blockIdx.x;  // global N-tile (0..127) in the [64,2048] GEMM
  int m  = mt * 16 + (lane & 15);
  int kh = (lane >> 4) & 1;

  v8f acc = {};
  // Recurrent part: h_in @ Whh^T, K = 512 -> 16 WMMAs
  {
    const _Float16* arow  = h_in + (size_t)m * HH + kh * 8;
    const _Float16* bbase = packWhh + (size_t)nt * 16 * 512 + lane * 16;
#pragma unroll 4
    for (int kt = 0; kt < 16; ++kt) {
      v16h a = frag_from_row(arow + kt * 32);
      v16h b = *(const v16h*)(bbase + (size_t)kt * 512);
      acc = __builtin_amdgcn_wmma_f32_16x16x32_f16(false, a, false, b, (short)0, acc, false, false);
    }
  }
  // Input part (encoder only): embedding gather (f32->f16) + K = 256 -> 8 WMMAs
  if (ids != nullptr) {
    const float* x = emb + (size_t)ids[m * LL + t] * EE + kh * 8;
    const _Float16* bbase = packWih + (size_t)nt * 8 * 512 + lane * 16;
#pragma unroll 2
    for (int kt = 0; kt < 8; ++kt) {
      const float* s = x + kt * 32;
      v16h a;
#pragma unroll
      for (int i = 0; i < 8; ++i) { a[i] = (_Float16)s[i]; a[i + 8] = (_Float16)s[16 + i]; }
      v16h b = *(const v16h*)(bbase + (size_t)kt * 512);
      acc = __builtin_amdgcn_wmma_f32_16x16x32_f16(false, a, false, b, (short)0, acc, false, false);
    }
  }
  // Stage gate tile in LDS. D layout: lane holds N=lane&15; VGPR v -> M=v+8*(lane>=16)
  int nl   = lane & 15;
  int mrow = mt * 16 + kh * 8;
  float bn = bias[q * HH + j0 + nl];
#pragma unroll
  for (int v = 0; v < 8; ++v)
    gbuf[q][mrow + v][nl] = acc[v] + bn;
  __syncthreads();

  // Cell update for this block's 64x16 slice (1024 elems, 512 threads x 2)
#pragma unroll
  for (int kidx = 0; kidx < 2; ++kidx) {
    int idx = threadIdx.x + kidx * 512;
    int b  = idx >> 4;
    int jj = idx & 15;
    float gi = gbuf[0][b][jj], gf = gbuf[1][b][jj];
    float gg = gbuf[2][b][jj], go = gbuf[3][b][jj];
    int gj = b * HH + j0 + jj;
    float cn = sigm(gf) * c[gj] + sigm(gi) * tanhf(gg);
    float h  = sigm(go) * tanhf(cn);
    c[gj]     = cn;
    h_out[gj] = (_Float16)h;
    if (enc_out) enc_out[gj] = (_Float16)h;
    if (h32_out) h32_out[gj] = h;
  }
}

// ---- Generic wave-per-tile f16 GEMM: C[M,N] = A[M,K] @ packB + bias ----
__global__ void k_gemm_f16(const _Float16* __restrict__ A, int lda, int KT,
                           const _Float16* __restrict__ packB, int Ntiles,
                           const float* __restrict__ bias,
                           float* __restrict__ C, int ldc) {
  int wave = blockIdx.x * (blockDim.x >> 5) + (threadIdx.x >> 5);
  int lane = threadIdx.x & 31;
  int mt = wave / Ntiles;
  int nt = wave % Ntiles;
  int m  = mt * 16 + (lane & 15);
  int kh = (lane >> 4) & 1;
  const _Float16* arow  = A + (size_t)m * lda + kh * 8;
  const _Float16* bbase = packB + (size_t)nt * KT * 512 + lane * 16;
  v8f acc = {};
  for (int kt = 0; kt < KT; ++kt) {
    v16h a = frag_from_row(arow + kt * 32);
    v16h b = *(const v16h*)(bbase + (size_t)kt * 512);
    acc = __builtin_amdgcn_wmma_f32_16x16x32_f16(false, a, false, b, (short)0, acc, false, false);
  }
  int n    = nt * 16 + (lane & 15);
  int mrow = mt * 16 + kh * 8;
  float bn = bias ? bias[n] : 0.0f;
#pragma unroll
  for (int v = 0; v < 8; ++v)
    C[(size_t)(mrow + v) * ldc + n] = acc[v] + bn;
}

// ---- Attention scores: sc[l*B+b] = sum_w tanh(blend1 + hW2) * vt + vt_b
// One wave per (l,b) row; blend1 is L2-resident, prefetch the row 8 ahead. ----
__global__ void k_scores(const float* __restrict__ blend1,  // [L*B, 256]
                         const float* __restrict__ hW2,     // [64, 256] (incl. W2_b)
                         const float* __restrict__ vt_w,
                         const float* __restrict__ vt_b,
                         float* __restrict__ sc) {           // [L*B]
  int wave = blockIdx.x * (blockDim.x >> 5) + (threadIdx.x >> 5);
  int lane = threadIdx.x & 31;
  int b = wave & 63;
  const float* row  = blend1 + (size_t)wave * WD;
  const float* hrow = hW2 + (size_t)b * WD;
  if (wave + 8 < LL * BB && lane < 16)
    __builtin_prefetch(row + 8 * WD + lane * 16, 0, 1);
  float acc = 0.0f;
#pragma unroll
  for (int i = 0; i < 8; ++i) {
    int w = lane + i * 32;
    acc += tanhf(row[w] + hrow[w]) * vt_w[w];
  }
#pragma unroll
  for (int off = 16; off > 0; off >>= 1) acc += __shfl_xor(acc, off, 32);
  if (lane == 0) sc[wave] = acc + vt_b[0];
}

// ---- log_softmax over L (axis 0) per batch column; scatter into out[L,B,T] ----
__global__ void k_logsoftmax(const float* __restrict__ sc, float* __restrict__ out, int t) {
  __shared__ float red[LL];
  int b = blockIdx.x;
  int l = threadIdx.x;
  float s = sc[(size_t)l * BB + b];
  red[l] = s;
  __syncthreads();
  for (int off = LL / 2; off > 0; off >>= 1) {
    if (l < off) red[l] = fmaxf(red[l], red[l + off]);
    __syncthreads();
  }
  float mx = red[0];
  __syncthreads();
  red[l] = __expf(s - mx);
  __syncthreads();
  for (int off = LL / 2; off > 0; off >>= 1) {
    if (l < off) red[l] += red[l + off];
    __syncthreads();
  }
  float lse = mx + __logf(red[0]);
  out[((size_t)l * BB + b) * TT + t] = s - lse;
}

extern "C" void kernel_launch(void* const* d_in, const int* in_sizes, int n_in,
                              void* d_out, int out_size, void* d_ws, size_t ws_size,
                              hipStream_t stream) {
  (void)in_sizes; (void)n_in; (void)out_size; (void)ws_size;
  const int*   ids    = (const int*)  d_in[0];
  const float* emb    = (const float*)d_in[1];
  const float* encWih = (const float*)d_in[2];
  const float* encWhh = (const float*)d_in[3];
  const float* encBih = (const float*)d_in[4];
  const float* encBhh = (const float*)d_in[5];
  // d_in[6] dec_Wih: dead (decoder input is all zeros)
  const float* decWhh = (const float*)d_in[7];
  const float* decBih = (const float*)d_in[8];
  const float* decBhh = (const float*)d_in[9];
  const float* W1w    = (const float*)d_in[10];
  const float* W1b    = (const float*)d_in[11];
  const float* W2w    = (const float*)d_in[12];
  const float* W2b    = (const float*)d_in[13];
  const float* vtw    = (const float*)d_in[14];
  const float* vtb    = (const float*)d_in[15];
  const float* h0     = (const float*)d_in[16];
  float* out = (float*)d_out;

  // Workspace carve (~71 MB; all L2-resident on MI455X's 192MB L2)
  char* p = (char*)d_ws;
  auto carve = [&](size_t bytes) { char* r = p; p += (bytes + 255) & ~(size_t)255; return r; };
  _Float16* pWihE = (_Float16*)carve((size_t)G4 * EE * 2);
  _Float16* pWhhE = (_Float16*)carve((size_t)G4 * HH * 2);
  _Float16* pWhhD = (_Float16*)carve((size_t)G4 * HH * 2);
  _Float16* pW1   = (_Float16*)carve((size_t)WD * HH * 2);
  _Float16* pW2   = (_Float16*)carve((size_t)WD * HH * 2);
  _Float16* encS  = (_Float16*)carve((size_t)LL * BB * HH * 2);   // 32 MB
  float*    blend1= (float*)   carve((size_t)LL * BB * WD * 4);   // 32 MB
  _Float16* hbufA = (_Float16*)carve((size_t)BB * HH * 2);
  _Float16* hbufB = (_Float16*)carve((size_t)BB * HH * 2);
  float*    cbuf  = (float*)   carve((size_t)BB * HH * 4);
  float*    cdec  = (float*)   carve((size_t)BB * HH * 4);
  float*    hW2   = (float*)   carve((size_t)BB * WD * 4);
  float*    sc    = (float*)   carve((size_t)LL * BB * 4);
  float*    ebias = (float*)   carve((size_t)G4 * 4);
  float*    dbias = (float*)   carve((size_t)G4 * 4);

  // One-time prep: pack weights to WMMA-B fragment order, fold biases
  k_pack_b<<<(G4 * EE) / 256, 256, 0, stream>>>(encWih, pWihE, G4, EE);
  k_pack_b<<<(G4 * HH) / 256, 256, 0, stream>>>(encWhh, pWhhE, G4, HH);
  k_pack_b<<<(G4 * HH) / 256, 256, 0, stream>>>(decWhh, pWhhD, G4, HH);
  k_pack_b<<<(WD * HH) / 256, 256, 0, stream>>>(W1w, pW1, WD, HH);
  k_pack_b<<<(WD * HH) / 256, 256, 0, stream>>>(W2w, pW2, WD, HH);
  k_add2<<<8, 256, 0, stream>>>(encBih, encBhh, ebias, G4);
  k_add2<<<8, 256, 0, stream>>>(decBih, decBhh, dbias, G4);
  hipMemsetAsync(hbufA, 0, (size_t)BB * HH * 2, stream);
  hipMemsetAsync(cbuf, 0, (size_t)BB * HH * 4, stream);

  // Encoder: 512 sequential steps, ONE fused kernel per step
  // (gather + 24 WMMAs per wave + LDS-staged gates + cell update).
  _Float16* hA = hbufA;
  _Float16* hB = hbufB;
  for (int t = 0; t < LL; ++t) {
    k_lstm_step<<<32, 512, 0, stream>>>(
        hA, hB, cbuf, emb, ids, t, pWihE, pWhhE, ebias,
        encS + (size_t)t * BB * HH,
        (t == LL - 1) ? cdec : nullptr);  // c0 = last encoder h
    _Float16* tmp = hA; hA = hB; hB = tmp;
  }

  // blend1 = enc_states @ W1^T + W1_b : 32768x512x256 WMMA GEMM (32768 waves)
  k_gemm_f16<<<8192, 128, 0, stream>>>(encS, HH, 16, pW1, WD / 16, W1b, blend1, WD);

  // Decoder init: h = h0 (f16 into hA), c = cdec (in place)
  k_f32_to_f16<<<(BB * HH) / 256, 256, 0, stream>>>(h0, hA, BB * HH);

  for (int t = 0; t < TT; ++t) {
    k_lstm_step<<<32, 512, 0, stream>>>(
        hA, hB, cdec, nullptr, nullptr, 0, nullptr, pWhhD, dbias, nullptr, nullptr);
    k_gemm_f16<<<16, 128, 0, stream>>>(hB, HH, 16, pW2, WD / 16, W2b, hW2, WD);
    k_scores<<<(LL * BB) / 8, 256, 0, stream>>>(blend1, hW2, vtw, vtb, sc);
    k_logsoftmax<<<BB, LL, 0, stream>>>(sc, out, t);
    _Float16* tmp = hA; hA = hB; hB = tmp;
  }
}
